// Attention_12000138625343
// MI455X (gfx1250) — matrix operand order repaired and checked by reading, not compiled, not run on hardware
//
#include <hip/hip_runtime.h>
#include <hip/hip_bf16.h>
#include <cstdint>
#include <math.h>

// ---- problem constants (match reference) ----
#define BSZ 2
#define SEQ 2048
#define EMB 1024
#define NH  16
#define HD  64
#define ROPE_BASE 10000.0f

typedef __attribute__((ext_vector_type(16))) __bf16 v16bf;
typedef __attribute__((ext_vector_type(8)))  __bf16 v8bf;
typedef __attribute__((ext_vector_type(8)))  float  v8f;

// ---------------------------------------------------------------------------
// CDNA5 async global->LDS copy (ASYNCcnt-tracked, no VGPR staging).
// ---------------------------------------------------------------------------
__device__ __forceinline__ void async_copy_b128(uint32_t lds_off, const void* g) {
    uint64_t ga = (uint64_t)(uintptr_t)g;
    asm volatile("global_load_async_to_lds_b128 %0, %1, off"
                 :: "v"(lds_off), "v"(ga) : "memory");
}
__device__ __forceinline__ void wait_async_le3() {
    asm volatile("s_wait_asynccnt 0x3" ::: "memory");
}
__device__ __forceinline__ void wait_async_le0() {
    asm volatile("s_wait_asynccnt 0x0" ::: "memory");
}

// Build a 16-element bf16 fragment from two aligned 16B chunks.
__device__ __forceinline__ v16bf frag16(const __bf16* p0, const __bf16* p1) {
    v8bf a = *(const v8bf*)p0;
    v8bf b = *(const v8bf*)p1;
    return __builtin_shufflevector(a, b, 0,1,2,3,4,5,6,7,8,9,10,11,12,13,14,15);
}

// ---------------------------------------------------------------------------
// f32 -> bf16 bulk conversion (4 elements / thread)
// ---------------------------------------------------------------------------
__global__ void __launch_bounds__(256)
cvt_f32_bf16(const float* __restrict__ src, __bf16* __restrict__ dst, size_t n) {
    size_t i = ((size_t)blockIdx.x * blockDim.x + threadIdx.x) * 4;
    if (i >= n) return;
    float4 v = *(const float4*)(src + i);
    dst[i + 0] = (__bf16)v.x;
    dst[i + 1] = (__bf16)v.y;
    dst[i + 2] = (__bf16)v.z;
    dst[i + 3] = (__bf16)v.w;
}

// ---------------------------------------------------------------------------
// GEMM: C[M][N] (f32) = A[M][K] (bf16) * Bt[N][K]^T (bf16)
// Workgroup = 8 waves -> 256x64 C tile; wave -> 64x32 (4x2 WMMA tiles).
// K-panels (A:256x32, B:64x32 bf16) double-buffered in LDS via async copies:
// prefetch k+1 while computing k, drained with s_wait_asynccnt <= 3
// (async loads complete in order, so <=3 retires the older group and 2 of the newer 5).
// ---------------------------------------------------------------------------
__global__ void __launch_bounds__(256)
gemm_bf16(const __bf16* __restrict__ A, const __bf16* __restrict__ Bt,
          float* __restrict__ C, int M, int N, int K) {
    __shared__ __bf16 sA[2][256 * 32];   // 2 x 16 KB
    __shared__ __bf16 sB[2][64 * 32];    // 2 x  4 KB
    const int tid  = threadIdx.x;
    const int lane = tid & 31;
    const int wave = tid >> 5;        // 0..7
    const int rw   = wave >> 1;       // 0..3  (M dir, 64 rows each)
    const int cw   = wave & 1;        // 0..1  (N dir, 32 cols each)
    const int l15  = lane & 15, hi = lane >> 4;
    const int bm   = blockIdx.x * 256;
    const int bn   = blockIdx.y * 64;

    v8f acc[4][2];
#pragma unroll
    for (int mi = 0; mi < 4; ++mi)
#pragma unroll
        for (int ni = 0; ni < 2; ++ni) acc[mi][ni] = (v8f){};

    const uint32_t ldsA = (uint32_t)(uintptr_t)&sA[0][0];
    const uint32_t ldsB = (uint32_t)(uintptr_t)&sB[0][0];

    // issue one k-panel's async copies into buffer p
    auto issue = [&](int p, int k0) {
        // A panel: 256 rows x 64B = 1024 chunks of 16B; 4 chunks / thread
#pragma unroll
        for (int j = 0; j < 4; ++j) {
            int c = tid + j * 256;
            async_copy_b128(ldsA + (uint32_t)(p * 16384 + (c >> 2) * 64 + (c & 3) * 16),
                            A + (size_t)(bm + (c >> 2)) * K + k0 + (c & 3) * 8);
        }
        // B panel: 64 rows x 64B = 256 chunks; 1 chunk / thread
        async_copy_b128(ldsB + (uint32_t)(p * 4096 + (tid >> 2) * 64 + (tid & 3) * 16),
                        Bt + (size_t)(bn + (tid >> 2)) * K + k0 + (tid & 3) * 8);
    };

    issue(0, 0);                 // prologue: prefetch first panel
    int p = 0;
    for (int k0 = 0; k0 < K; k0 += 32) {
        const bool more = (k0 + 32) < K;
        if (more) issue(p ^ 1, k0 + 32);        // prefetch next panel
        if (more) wait_async_le3();             // drain older group of 5 plus 2 of the newer
        else      wait_async_le0();
        __syncthreads();                        // panel p visible to all waves

        v16bf af[4], bfr[2];
#pragma unroll
        for (int mi = 0; mi < 4; ++mi) {
            const __bf16* q = &sA[p][(rw * 64 + mi * 16 + l15) * 32 + hi * 16];
            af[mi] = frag16(q, q + 8);
        }
#pragma unroll
        for (int ni = 0; ni < 2; ++ni) {
            const __bf16* q = &sB[p][(cw * 32 + ni * 16 + l15) * 32 + hi * 16];
            bfr[ni] = frag16(q, q + 8);
        }
#pragma unroll
        for (int mi = 0; mi < 4; ++mi)
#pragma unroll
            for (int ni = 0; ni < 2; ++ni)
                acc[mi][ni] = __builtin_amdgcn_wmma_f32_16x16x32_bf16(
                    false, af[mi], false, bfr[ni], (short)0, acc[mi][ni], false, false);

        __syncthreads();                        // all reads of p done before overwrite
        p ^= 1;
    }

#pragma unroll
    for (int mi = 0; mi < 4; ++mi)
#pragma unroll
        for (int ni = 0; ni < 2; ++ni) {
            int m00 = bm + rw * 64 + mi * 16;
            int n00 = bn + cw * 32 + ni * 16;
#pragma unroll
            for (int r = 0; r < 8; ++r)
                C[(size_t)(m00 + r + hi * 8) * N + n00 + l15] = acc[mi][ni][r];
        }
}

// ---------------------------------------------------------------------------
// RoPE (interleaved) on q,k + pass-through v, f32 qkv -> bf16 qkvb
// qkv layout [B][S][3][NH][HD]
// ---------------------------------------------------------------------------
__global__ void __launch_bounds__(256)
rope_cvt_kernel(const float* __restrict__ qkv, __bf16* __restrict__ qkvb) {
    const int HALF = HD / 2;  // 32 pairs
    size_t idx = (size_t)blockIdx.x * blockDim.x + threadIdx.x;
    const size_t total = (size_t)BSZ * SEQ * 3 * NH * HALF;
    if (idx >= total) return;

    int p  = (int)(idx % HALF); idx /= HALF;
    int h  = (int)(idx % NH);   idx /= NH;
    int i3 = (int)(idx % 3);    idx /= 3;
    int s  = (int)(idx % SEQ);
    int b  = (int)(idx / SEQ);

    size_t off = ((((size_t)b * SEQ + s) * 3 + i3) * NH + h) * HD + 2 * p;
    float x0 = qkv[off], x1 = qkv[off + 1];
    if (i3 < 2) {  // rotate q and k only
        float theta = __powf(ROPE_BASE, -(float)p / (float)HALF);
        float ang = (float)s * theta;
        float c = __cosf(ang), sn = __sinf(ang);
        float r0 = x0 * c - x1 * sn;
        float r1 = x1 * c + x0 * sn;
        x0 = r0; x1 = r1;
    }
    qkvb[off]     = (__bf16)x0;
    qkvb[off + 1] = (__bf16)x1;
}

// ---------------------------------------------------------------------------
// Flash attention over bf16 qkv: one wave per 16-row q block per (b,h), Bc=32.
// 1/sqrt(HD) applied to f32 scores. Output written as bf16.
// ---------------------------------------------------------------------------
__global__ void __launch_bounds__(128)
attn_kernel(const __bf16* __restrict__ qkvb, __bf16* __restrict__ obf) {
    const int lane  = threadIdx.x & 31;
    const int wslot = threadIdx.x >> 5;
    const int wave  = blockIdx.x * 4 + wslot;
    const int QB    = SEQ / 16;
    const int qblk  = wave % QB;
    const int bh    = wave / QB;
    const int h     = bh % NH;
    const int b     = bh / NH;
    const int l15   = lane & 15, hi = lane >> 4;
    const int q0    = qblk * 16;

    const size_t rs = 3 * (size_t)EMB;  // seq stride (elements)
    const __bf16* qbase = qkvb + ((size_t)b * SEQ) * rs + 0 * EMB + h * HD;
    const __bf16* kbase = qkvb + ((size_t)b * SEQ) * rs + 1 * EMB + h * HD;
    const __bf16* vbase = qkvb + ((size_t)b * SEQ) * rs + 2 * EMB + h * HD;

    v16bf aq[2];
#pragma unroll
    for (int kc = 0; kc < 2; ++kc) {
        const __bf16* p = qbase + (size_t)(q0 + l15) * rs + kc * 32 + hi * 16;
        aq[kc] = frag16(p, p + 8);
    }

    float mrow[8], lrow[8];
    v8f oacc[4];
#pragma unroll
    for (int r = 0; r < 8; ++r) { mrow[r] = -3.0e38f; lrow[r] = 0.0f; }
#pragma unroll
    for (int t = 0; t < 4; ++t) oacc[t] = (v8f){};

    __shared__ __bf16 pls[4][16][32];

    for (int j0 = 0; j0 < SEQ; j0 += 32) {
        // ---- scores S = q k^T (16 x 32) ----
        v8f s0 = {}, s1 = {};
        {
            const __bf16* kp0 = kbase + (size_t)(j0 + l15) * rs + hi * 16;
            const __bf16* kp1 = kbase + (size_t)(j0 + 16 + l15) * rs + hi * 16;
            v16bf b00 = frag16(kp0,      kp0 + 8);
            v16bf b01 = frag16(kp0 + 32, kp0 + 40);
            v16bf b10 = frag16(kp1,      kp1 + 8);
            v16bf b11 = frag16(kp1 + 32, kp1 + 40);
            s0 = __builtin_amdgcn_wmma_f32_16x16x32_bf16(false, aq[0], false, b00, (short)0, s0, false, false);
            s0 = __builtin_amdgcn_wmma_f32_16x16x32_bf16(false, aq[1], false, b01, (short)0, s0, false, false);
            s1 = __builtin_amdgcn_wmma_f32_16x16x32_bf16(false, aq[0], false, b10, (short)0, s1, false, false);
            s1 = __builtin_amdgcn_wmma_f32_16x16x32_bf16(false, aq[1], false, b11, (short)0, s1, false, false);
        }

        // ---- online softmax (scale folded here) ----
        float p0a[8], p1a[8], alpha[8];
#pragma unroll
        for (int r = 0; r < 8; ++r) {
            float a0 = s0[r] * 0.125f, a1 = s1[r] * 0.125f;
            float mx = fmaxf(a0, a1);
            mx = fmaxf(mx, __shfl_xor(mx, 1, 16));
            mx = fmaxf(mx, __shfl_xor(mx, 2, 16));
            mx = fmaxf(mx, __shfl_xor(mx, 4, 16));
            mx = fmaxf(mx, __shfl_xor(mx, 8, 16));
            float mn = fmaxf(mrow[r], mx);
            float al = __expf(mrow[r] - mn);
            float p0 = __expf(a0 - mn);
            float p1 = __expf(a1 - mn);
            float sum = p0 + p1;
            sum += __shfl_xor(sum, 1, 16);
            sum += __shfl_xor(sum, 2, 16);
            sum += __shfl_xor(sum, 4, 16);
            sum += __shfl_xor(sum, 8, 16);
            lrow[r] = lrow[r] * al + sum;
            mrow[r] = mn;
            alpha[r] = al; p0a[r] = p0; p1a[r] = p1;
        }
#pragma unroll
        for (int t = 0; t < 4; ++t)
#pragma unroll
            for (int r = 0; r < 8; ++r) oacc[t][r] *= alpha[r];

        // ---- P: C-layout -> A-layout via per-wave LDS ----
        __syncthreads();
#pragma unroll
        for (int r = 0; r < 8; ++r) {
            pls[wslot][r + hi * 8][l15]      = (__bf16)p0a[r];
            pls[wslot][r + hi * 8][16 + l15] = (__bf16)p1a[r];
        }
        __syncthreads();
        const __bf16* pp = &pls[wslot][l15][hi * 16];
        v16bf ap = frag16(pp, pp + 8);

        // ---- O += P @ V ----
        const __bf16* vrow = vbase + (size_t)(j0 + hi * 16) * rs;
#pragma unroll
        for (int t = 0; t < 4; ++t) {
            v16bf bv;
            const __bf16* vp = vrow + 16 * t + l15;
#pragma unroll
            for (int i = 0; i < 16; ++i) bv[i] = vp[(size_t)i * rs];
            oacc[t] = __builtin_amdgcn_wmma_f32_16x16x32_bf16(
                false, ap, false, bv, (short)0, oacc[t], false, false);
        }
    }

    // ---- normalize, store O as bf16 [b][s][h*HD+d] ----
    __bf16* orow = obf + ((size_t)b * SEQ) * EMB + h * HD;
#pragma unroll
    for (int r = 0; r < 8; ++r) {
        float inv = 1.0f / lrow[r];
        int srow = q0 + r + hi * 8;
#pragma unroll
        for (int t = 0; t < 4; ++t)
            orow[(size_t)srow * EMB + 16 * t + l15] = (__bf16)(oacc[t][r] * inv);
    }
}

// ---------------------------------------------------------------------------
extern "C" void kernel_launch(void* const* d_in, const int* in_sizes, int n_in,
                              void* d_out, int out_size, void* d_ws, size_t ws_size,
                              hipStream_t stream) {
    (void)in_sizes; (void)n_in; (void)out_size; (void)ws_size;
    const float* x  = (const float*)d_in[0];   // [B,S,E]
    const float* w1 = (const float*)d_in[1];   // [3E,E]
    const float* w2 = (const float*)d_in[2];   // [E,E]
    float* out = (float*)d_out;                // [B,S,E]

    // workspace carve-up (bytes)
    char* ws = (char*)d_ws;
    const size_t M = (size_t)BSZ * SEQ;        // 4096
    float*  qkv   = (float*)ws;                              ws += M * 3 * EMB * 4;  // 48 MB
    __bf16* qkvb  = (__bf16*)ws;                             ws += M * 3 * EMB * 2;  // 24 MB
    __bf16* xb    = (__bf16*)ws;                             ws += M * EMB * 2;      // 8 MB
    __bf16* w1b   = (__bf16*)ws;                             ws += (size_t)3 * EMB * EMB * 2;
    __bf16* w2b   = (__bf16*)ws;                             ws += (size_t)EMB * EMB * 2;
    __bf16* ob    = (__bf16*)ws;

    // 0) one-time bf16 shadows of x, w1, w2
    {
        size_t nx = M * EMB, n1 = (size_t)3 * EMB * EMB, n2 = (size_t)EMB * EMB;
        cvt_f32_bf16<<<dim3((unsigned)(nx / 4 / 256)), dim3(256), 0, stream>>>(x, xb, nx);
        cvt_f32_bf16<<<dim3((unsigned)(n1 / 4 / 256)), dim3(256), 0, stream>>>(w1, w1b, n1);
        cvt_f32_bf16<<<dim3((unsigned)(n2 / 4 / 256)), dim3(256), 0, stream>>>(w2, w2b, n2);
    }
    // 1) qkv = x @ w1^T   (f32 out, for RoPE precision)
    gemm_bf16<<<dim3(16, 48), dim3(256), 0, stream>>>(xb, w1b, qkv, (int)M, 3 * EMB, EMB);
    // 2) RoPE q,k + bf16 conversion of whole qkv
    {
        size_t total = (size_t)BSZ * SEQ * 3 * NH * (HD / 2);
        rope_cvt_kernel<<<dim3((unsigned)((total + 255) / 256)), dim3(256), 0, stream>>>(qkv, qkvb);
    }
    // 3) flash attention -> bf16 O
    attn_kernel<<<dim3(BSZ * NH * (SEQ / 16) / 4), dim3(128), 0, stream>>>(qkvb, ob);
    // 4) out = O @ w2^T
    gemm_bf16<<<dim3(16, 16), dim3(256), 0, stream>>>(ob, w2b, out, (int)M, EMB, EMB);
}
